// InverseNet_for_STL_1941325217900
// MI455X (gfx1250) — matrix operand order repaired and checked by reading, not compiled, not run on hardware
//
#include <hip/hip_runtime.h>
#include <hip/hip_bf16.h>
#include <math.h>

// ---------------------------------------------------------------------------
// CDNA5 (gfx1250) implementation of the InverseNet FISTA encoder/decoder.
// All conv / conv_transpose work is implicit GEMM on the WMMA pipe
// (v_wmma_f32_16x16x32_bf16, f32 accumulate). Weight tiles are staged with
// GLOBAL_LOAD_ASYNC_TO_LDS_B128 (ASYNCcnt), activations are im2col-gathered
// with geometry as compile-time constants (shifts, no idiv).
// ---------------------------------------------------------------------------

typedef __attribute__((ext_vector_type(16))) __bf16 v16bf;
typedef __attribute__((ext_vector_type(8)))  __bf16 v8bf;
typedef __attribute__((ext_vector_type(8)))  float  v8f;

#define LMBD 0.1f
#define MU   0.1f
#define THR  (MU * LMBD)

__device__ __forceinline__ void async_wait0() {
#if __has_builtin(__builtin_amdgcn_s_wait_asynccnt)
  __builtin_amdgcn_s_wait_asynccnt(0);
#else
  asm volatile("s_wait_asynccnt 0x0" ::: "memory");
#endif
}

// ------------------------- small helper kernels ----------------------------

__global__ void wnorm_kernel(const float* __restrict__ W, int len,
                             float* __restrict__ invn) {
  __shared__ float red[256];
  const int co = blockIdx.x;
  const float* w = W + (size_t)co * len;
  float s = 0.f;
  for (int i = threadIdx.x; i < len; i += 256) { float v = w[i]; s += v * v; }
  red[threadIdx.x] = s;
  __syncthreads();
  for (int k = 128; k > 0; k >>= 1) {
    if (threadIdx.x < k) red[threadIdx.x] += red[threadIdx.x + k];
    __syncthreads();
  }
  if (threadIdx.x == 0) invn[co] = 1.0f / (sqrtf(red[0]) + 1e-12f);
}

// Pack forward-conv weights as GEMM A: M=Co, k=(ci,kh,kw), bf16,
// layout [k/32][Mpad][32] (zero-padded).
__global__ void pack_fwd_kernel(const float* __restrict__ W,
                                const float* __restrict__ invn,
                                __bf16* __restrict__ Ap,
                                int Co, int CiKHKW, int Kpad, int Mpad) {
  int idx = blockIdx.x * 256 + threadIdx.x;
  if (idx >= Mpad * Kpad) return;
  int m = idx / Kpad, k = idx % Kpad;
  float v = 0.f;
  if (m < Co && k < CiKHKW) v = W[(size_t)m * CiKHKW + k] * invn[m];
  Ap[((size_t)(k >> 5) * Mpad + m) * 32 + (k & 31)] = (__bf16)v;
}

// Pack conv-transpose weights as GEMM A: M=Ci, k=(co,kh,kw).
__global__ void pack_bwd_kernel(const float* __restrict__ W,
                                const float* __restrict__ invn,
                                __bf16* __restrict__ Ap,
                                int Co, int Ci, int KHKW, int Kpad, int Mpad) {
  int idx = blockIdx.x * 256 + threadIdx.x;
  if (idx >= Mpad * Kpad) return;
  int m = idx / Kpad, k = idx % Kpad;
  float v = 0.f;
  if (m < Ci && k < Co * KHKW) {
    int co = k / KHKW, r = k % KHKW;
    v = W[((size_t)co * Ci + m) * KHKW + r] * invn[co];
  }
  Ap[((size_t)(k >> 5) * Mpad + m) * 32 + (k & 31)] = (__bf16)v;
}

// ----------------------- implicit-GEMM WMMA kernel -------------------------
// C[M,N] = A[M,K] * B[K,N], B gathered on the fly (im2col / im2colT),
// epilogue: out = f(alpha*acc + bias + add), f = relu or identity.
// Tile 64(M) x 256(N) x 32(K), 256 threads = 8 wave32.
// Per wave: 64x32 strip = 4 A-frags x 2 B-frags = 8 WMMA per k-step.
//   MODE: 0 = conv, 1 = convT
//   CFG:  0 = k4 s2 p1 (KHKW=16, channel constant per 16-k half-block)
//         1 = k3 s1 p0 (KHKW=9, constant-divisor generic path)
template <int MODE, int CFG>
__global__ __launch_bounds__(256)
void gemm_conv_kernel(const __bf16* __restrict__ Ap,
                      const float* __restrict__ src,
                      const float* __restrict__ add,
                      float* __restrict__ out,
                      int M, int N, int Kpad, int Kreal, int Mpad,
                      int Cs, int Hs, int Ws_,        // source tensor dims
                      int OH, int OW,                 // output spatial dims
                      float alpha, float bias, int relu) {
  constexpr int KH = (CFG == 0) ? 4 : 3;
  constexpr int KW = KH;
  constexpr int KHKW = KH * KW;
  constexpr int STRIDE = (CFG == 0) ? 2 : 1;
  constexpr int PAD    = (CFG == 0) ? 1 : 0;

  __shared__ __bf16 As[2][64 * 32];
  __shared__ __bf16 Bs[2][256 * 32];

  const int tid  = threadIdx.x;
  const int wave = tid >> 5;
  const int lane = tid & 31;
  const int m0 = blockIdx.y * 64;
  const int n0 = blockIdx.x * 256;
  const int OHW = OH * OW;
  const int nk = Kpad >> 5;

  v8f acc[2][4];
#pragma unroll
  for (int c2 = 0; c2 < 2; ++c2)
#pragma unroll
    for (int r = 0; r < 4; ++r) acc[c2][r] = {};

  // B-gather column coordinates: one full 32-K column per thread.
  const int bn = n0 + tid;
  const bool col_ok = (bn < N);
  int nb = 0, oy = 0, ox = 0;
  if (col_ok) {
    nb = bn / OHW;
    int pix = bn - nb * OHW;
    oy = pix / OW;
    ox = pix - oy * OW;
  }
  const int arow = tid >> 2, achunk = tid & 3;

  auto pk2 = [](float a, float b) -> unsigned int {
    union { __bf16 h[2]; unsigned int u; } t;
    t.h[0] = (__bf16)a; t.h[1] = (__bf16)b;
    return t.u;
  };

  auto loadTile = [&](int kb, int buf) {
    // --- A tile: 64 rows x 32 bf16; one async 16B copy per lane ---
    {
      const v8bf* g =
          (const v8bf*)Ap + (size_t)(kb * Mpad + m0 + arow) * 4 + achunk;
      unsigned lds = (unsigned)(size_t)&As[buf][arow * 32 + achunk * 8];
      asm volatile("global_load_async_to_lds_b128 %0, %1, off"
                   :: "v"(lds), "v"((unsigned long long)(size_t)g)
                   : "memory");
    }
    // --- B tile: this thread's column, 32 K values in two 16-k halves ---
#pragma unroll
    for (int half = 0; half < 2; ++half) {
      float vals[16];
#pragma unroll
      for (int i = 0; i < 16; ++i) vals[i] = 0.f;

      const int kblk = kb * 32 + half * 16;
      if constexpr (CFG == 0) {
        // KHKW == 16: the 16-k half-block maps to exactly one channel.
        if (col_ok && kblk < Kreal) {
          const int c = kblk >> 4;
          const float* srcc = src + ((size_t)nb * Cs + c) * Hs * Ws_;
#pragma unroll
          for (int kh = 0; kh < KH; ++kh) {
            int hrow; bool row_ok;
            if constexpr (MODE == 0) {
              hrow = oy * STRIDE - PAD + kh;
              row_ok = (hrow >= 0) & (hrow < Hs);
            } else {
              int hn = oy + PAD - kh;
              hrow = hn >> 1;
              row_ok = (hn >= 0) & ((hn & 1) == 0) & (hrow < Hs);
            }
            const float* rowp = srcc + (size_t)(row_ok ? hrow : 0) * Ws_;
#pragma unroll
            for (int kw = 0; kw < KW; ++kw) {
              float v = 0.f;
              if constexpr (MODE == 0) {
                int wi = ox * STRIDE - PAD + kw;
                if (row_ok && wi >= 0 && wi < Ws_) v = rowp[wi];
              } else {
                int wn = ox + PAD - kw;
                int wo = wn >> 1;
                if (row_ok && wn >= 0 && ((wn & 1) == 0) && wo < Ws_)
                  v = rowp[wo];
              }
              vals[kh * KW + kw] = v;
            }
          }
        }
      } else {
        // KHKW == 9: constant-divisor path (strength-reduced /9, /3).
#pragma unroll
        for (int i = 0; i < 16; ++i) {
          int k = kblk + i;
          float v = 0.f;
          if (col_ok && k < Kreal) {
            int c = k / KHKW, r = k - c * KHKW;
            int kh = r / KW, kw = r - kh * KW;
            if constexpr (MODE == 0) {
              int hi = oy + kh, wi = ox + kw;  // stride 1, pad 0
              if (hi < Hs && wi < Ws_)
                v = src[(((size_t)nb * Cs + c) * Hs + hi) * Ws_ + wi];
            } else {
              int ho = oy - kh, wo = ox - kw;
              if (ho >= 0 && ho < Hs && wo >= 0 && wo < Ws_)
                v = src[(((size_t)nb * Cs + c) * Hs + ho) * Ws_ + wo];
            }
          }
          vals[i] = v;
        }
      }
      // Pack 16 bf16 into two 16B LDS stores.
      uint4 q0, q1;
      q0.x = pk2(vals[0], vals[1]);   q0.y = pk2(vals[2], vals[3]);
      q0.z = pk2(vals[4], vals[5]);   q0.w = pk2(vals[6], vals[7]);
      q1.x = pk2(vals[8], vals[9]);   q1.y = pk2(vals[10], vals[11]);
      q1.z = pk2(vals[12], vals[13]); q1.w = pk2(vals[14], vals[15]);
      uint4* bp = (uint4*)&Bs[buf][tid * 32 + half * 16];
      bp[0] = q0;
      bp[1] = q1;
    }
  };

  loadTile(0, 0);
  async_wait0();
  __syncthreads();

  for (int kb = 0; kb < nk; ++kb) {
    const int cur = kb & 1;
    if (kb + 1 < nk) {
      __builtin_prefetch(
          (const char*)((const v8bf*)Ap + (size_t)((kb + 1) * Mpad + m0) * 4),
          0, 0);
      loadTile(kb + 1, cur ^ 1);
    }

    // Two B fragments for this wave's 32-column strip (ISA 7.12.2 layout):
    // lanes 0-15 hold K {0..7,16..23}, lanes 16-31 hold K {8..15,24..31}.
    v16bf bf[2];
    {
      const v8bf* bp = (const v8bf*)Bs[cur];
#pragma unroll
      for (int c2 = 0; c2 < 2; ++c2) {
        int col  = wave * 32 + c2 * 16 + (lane & 15);
        int base = col * 4 + ((lane < 16) ? 0 : 1);
        v8bf lo = bp[base];
        v8bf hi = bp[base + 2];
        for (int i = 0; i < 8; ++i) { bf[c2][i] = lo[i]; bf[c2][i + 8] = hi[i]; }
      }
    }

    const v8bf* apk = (const v8bf*)As[cur];
#pragma unroll
    for (int r = 0; r < 4; ++r) {
      // A fragment: lane<16 -> K {0..7,16..23}; lane>=16 -> K {8..15,24..31}.
      int row  = r * 16 + (lane & 15);
      int base = row * 4 + ((lane < 16) ? 0 : 1);
      v8bf lo = apk[base];
      v8bf hi = apk[base + 2];
      v16bf af;
      for (int i = 0; i < 8; ++i) { af[i] = lo[i]; af[i + 8] = hi[i]; }
      acc[0][r] = __builtin_amdgcn_wmma_f32_16x16x32_bf16(
          false, af, false, bf[0], (short)0, acc[0][r], false, false);
      acc[1][r] = __builtin_amdgcn_wmma_f32_16x16x32_bf16(
          false, af, false, bf[1], (short)0, acc[1][r], false, false);
    }
    async_wait0();
    __syncthreads();
  }

  // Epilogue (C layout: VGPR j -> M = j + (lane>=16 ? 8 : 0), N = lane&15).
  const int moff = (lane >= 16) ? 8 : 0;
#pragma unroll
  for (int c2 = 0; c2 < 2; ++c2) {
    const int ncol = n0 + wave * 32 + c2 * 16 + (lane & 15);
    if (ncol < N) {
      const int onb  = ncol / OHW;
      const int opix = ncol - onb * OHW;
#pragma unroll
      for (int r = 0; r < 4; ++r) {
        for (int j = 0; j < 8; ++j) {
          int m = m0 + r * 16 + moff + j;
          if (m < M) {
            size_t idx = ((size_t)onb * M + m) * OHW + opix;
            float v = alpha * acc[c2][r][j] + bias;
            if (add) v += add[idx];
            if (relu) v = v > 0.f ? v : 0.f;
            out[idx] = v;
          }
        }
      }
    }
  }
}

// ----------------------------- pointwise ops -------------------------------

__global__ void bn_stats_kernel(const float* __restrict__ x, int Bn, int C,
                                int HW, float* __restrict__ bnm,
                                float* __restrict__ bni) {
  __shared__ float s1[256], s2[256];
  const int c = blockIdx.x;
  const int cnt = Bn * HW;
  float a = 0.f, b = 0.f;
  for (int i = threadIdx.x; i < cnt; i += 256) {
    int nb = i / HW, pix = i - nb * HW;
    float v = x[((size_t)nb * C + c) * HW + pix];
    a += v;
    b += v * v;
  }
  s1[threadIdx.x] = a; s2[threadIdx.x] = b;
  __syncthreads();
  for (int k = 128; k > 0; k >>= 1) {
    if (threadIdx.x < k) {
      s1[threadIdx.x] += s1[threadIdx.x + k];
      s2[threadIdx.x] += s2[threadIdx.x + k];
    }
    __syncthreads();
  }
  if (threadIdx.x == 0) {
    float m = s1[0] / cnt;
    float var = s2[0] / cnt - m * m;
    bnm[c] = m;
    bni[c] = rsqrtf(var + 1e-5f);
  }
}

// optional BN, then leaky-relu (leak=0.2) or relu (leak=0).
__global__ void act_kernel(const float* __restrict__ x, float* __restrict__ y,
                           int n, int C, int HW, const float* __restrict__ bnm,
                           const float* __restrict__ bni, int use_bn,
                           float leak) {
  int i = blockIdx.x * 256 + threadIdx.x;
  if (i >= n) return;
  float v = x[i];
  if (use_bn) {
    int c = (i / HW) % C;
    v = (v - bnm[c]) * bni[c];
  }
  y[i] = v >= 0.f ? v : leak * v;
}

__global__ void fista_y_kernel(const float* __restrict__ z,
                               const float* __restrict__ zo,
                               float* __restrict__ y, float c, int n) {
  int i = blockIdx.x * 256 + threadIdx.x;
  if (i >= n) return;
  float zv = z[i];
  y[i] = zv + c * (zv - zo[i]);
}

__global__ void tanh_kernel(const float* __restrict__ x, float* __restrict__ y,
                            int n) {
  int i = blockIdx.x * 256 + threadIdx.x;
  if (i >= n) return;
  y[i] = tanhf(x[i]);
}

// row-wise L2 normalize (B rows of 128), writes to d_out and decoder input.
__global__ void norm_kernel(const float* __restrict__ z,
                            float* __restrict__ outz,
                            float* __restrict__ zdec, int nz) {
  __shared__ float red[128];
  const int r = blockIdx.x, t = threadIdx.x;
  float v = z[(size_t)r * nz + t];
  red[t] = v * v;
  __syncthreads();
  for (int k = 64; k > 0; k >>= 1) {
    if (t < k) red[t] += red[t + k];
    __syncthreads();
  }
  float nrm = sqrtf(red[0]);
  nrm = nrm > 1e-12f ? nrm : 1e-12f;
  float o = v / nrm;
  outz[(size_t)r * nz + t] = o;
  zdec[(size_t)r * nz + t] = o;
}

// ------------------------------ host driver --------------------------------

extern "C" void kernel_launch(void* const* d_in, const int* in_sizes, int n_in,
                              void* d_out, int out_size, void* d_ws,
                              size_t ws_size, hipStream_t stream) {
  (void)in_sizes; (void)n_in; (void)out_size; (void)ws_size;

  const float* x = (const float*)d_in[0];
  const float* W[6];
  for (int i = 0; i < 6; ++i) W[i] = (const float*)d_in[1 + i];
  float* out = (float*)d_out;

  struct LD { int Ci, Co, Hi, Wi, Ho, Wo, K, s, p; };
  const LD L[6] = {
      {3,    64,   96, 96, 48, 48, 4, 2, 1},
      {64,   128,  48, 48, 24, 24, 4, 2, 1},
      {128,  256,  24, 24, 12, 12, 4, 2, 1},
      {256,  512,  12, 12, 6,  6,  4, 2, 1},
      {512,  1024, 6,  6,  3,  3,  4, 2, 1},
      {1024, 128,  3,  3,  1,  1,  3, 1, 0}};
  const int Bn = 64;

  // ---- workspace bump allocator ----
  size_t off = 0;
  auto alloc = [&](size_t bytes) -> void* {
    void* p = (char*)d_ws + off;
    off += (bytes + 255) & ~(size_t)255;
    return p;
  };
  const size_t MAXF = 9437184;  // 64*64*48*48 (largest activation)
  float* bufA  = (float*)alloc(MAXF * 4);
  float* bufZ  = (float*)alloc(MAXF * 4);
  float* bufZo = (float*)alloc(MAXF * 4);
  float* bufY  = (float*)alloc(MAXF * 4);
  float* bufR  = (float*)alloc(MAXF * 4);
  __bf16* Apf = (__bf16*)alloc((size_t)8388608 * 2);
  __bf16* Apb = (__bf16*)alloc((size_t)8388608 * 2);
  float* invn = (float*)alloc(1024 * 4);
  float* bnm  = (float*)alloc(1024 * 4);
  float* bni  = (float*)alloc(1024 * 4);
  float* zdec = (float*)alloc(8192 * 4);

  auto ru = [](int v, int a) { return (v + a - 1) / a * a; };

  auto gemm = [&](const __bf16* Ap, const float* src, const float* add,
                  float* o, int M, int N, int Kpad, int Kreal, int Mpad,
                  int mode, int Cs, int Hs, int Ws2, int OH, int OW, int Ksz,
                  float alpha, float bias, int relu) {
    dim3 g((N + 255) / 256, Mpad / 64);
    if (Ksz == 4) {
      if (mode == 0)
        gemm_conv_kernel<0, 0><<<g, 256, 0, stream>>>(
            Ap, src, add, o, M, N, Kpad, Kreal, Mpad, Cs, Hs, Ws2, OH, OW,
            alpha, bias, relu);
      else
        gemm_conv_kernel<1, 0><<<g, 256, 0, stream>>>(
            Ap, src, add, o, M, N, Kpad, Kreal, Mpad, Cs, Hs, Ws2, OH, OW,
            alpha, bias, relu);
    } else {
      if (mode == 0)
        gemm_conv_kernel<0, 1><<<g, 256, 0, stream>>>(
            Ap, src, add, o, M, N, Kpad, Kreal, Mpad, Cs, Hs, Ws2, OH, OW,
            alpha, bias, relu);
      else
        gemm_conv_kernel<1, 1><<<g, 256, 0, stream>>>(
            Ap, src, add, o, M, N, Kpad, Kreal, Mpad, Cs, Hs, Ws2, OH, OW,
            alpha, bias, relu);
    }
  };

  // ------------------------------ encoder ------------------------------
  const float* xin = x;
  float* Z  = bufZ;
  float* Zo = bufZo;

  for (int l = 0; l < 6; ++l) {
    const LD& d = L[l];
    const int KK  = d.K * d.K;
    const int Kf  = d.Ci * KK, Kfp = ru(Kf, 32), Mfp = ru(d.Co, 64);
    const int Kb  = d.Co * KK, Kbp = ru(Kb, 32), Mbp = ru(d.Ci, 64);
    const int Nf  = Bn * d.Ho * d.Wo;   // GEMM cols, forward conv
    const int Nbk = Bn * d.Hi * d.Wi;   // GEMM cols, conv transpose
    const int nz  = Bn * d.Co * d.Ho * d.Wo;

    wnorm_kernel<<<d.Co, 256, 0, stream>>>(W[l], Kf, invn);
    pack_fwd_kernel<<<(Mfp * Kfp + 255) / 256, 256, 0, stream>>>(
        W[l], invn, Apf, d.Co, Kf, Kfp, Mfp);
    pack_bwd_kernel<<<(Mbp * Kbp + 255) / 256, 256, 0, stream>>>(
        W[l], invn, Apb, d.Co, d.Ci, KK, Kbp, Mbp);

    // z0 = shrink(MU * conv(x, W))
    gemm(Apf, xin, nullptr, Z, d.Co, Nf, Kfp, Kf, Mfp, 0, d.Ci, d.Hi, d.Wi,
         d.Ho, d.Wo, d.K, MU, -THR, 1);
    hipMemcpyAsync(Zo, Z, (size_t)nz * 4, hipMemcpyDeviceToDevice, stream);

    double t = 1.0;
    for (int it = 0; it < 3; ++it) {  // N_STEPS - 1 FISTA iterations
      double tn = (1.0 + sqrt(1.0 + 4.0 * t * t)) * 0.5;
      float c = (float)((t - 1.0) / tn);
      t = tn;
      // y = z + c*(z - z_old)
      fista_y_kernel<<<(nz + 255) / 256, 256, 0, stream>>>(Z, Zo, bufY, c, nz);
      // r = x - convT(y)
      gemm(Apb, bufY, xin, bufR, d.Ci, Nbk, Kbp, Kb, Mbp, 1, d.Co, d.Ho, d.Wo,
           d.Hi, d.Wi, d.K, -1.f, 0.f, 0);
      // z_new = shrink(y + MU * conv(r))  -> into Zo, then swap
      gemm(Apf, bufR, bufY, Zo, d.Co, Nf, Kfp, Kf, Mfp, 0, d.Ci, d.Hi, d.Wi,
           d.Ho, d.Wo, d.K, MU, -THR, 1);
      float* tmp = Z; Z = Zo; Zo = tmp;
    }

    if (l < 5) {
      const int HW = d.Ho * d.Wo;
      if (l == 0) {
        act_kernel<<<(nz + 255) / 256, 256, 0, stream>>>(
            Z, bufA, nz, d.Co, HW, nullptr, nullptr, 0, 0.2f);
      } else {
        bn_stats_kernel<<<d.Co, 256, 0, stream>>>(Z, Bn, d.Co, HW, bnm, bni);
        act_kernel<<<(nz + 255) / 256, 256, 0, stream>>>(
            Z, bufA, nz, d.Co, HW, bnm, bni, 1, 0.2f);
      }
      xin = bufA;
    }
  }

  // z normalize -> d_out[0:8192) and decoder input
  norm_kernel<<<Bn, 128, 0, stream>>>(Z, out, zdec, 128);

  // ------------------------------ decoder ------------------------------
  float* P0 = bufZ;
  float* P1 = bufY;
  const float* h = zdec;

  for (int l = 5; l >= 0; --l) {
    const LD& d = L[l];
    const int KK  = d.K * d.K;
    const int Kb  = d.Co * KK, Kbp = ru(Kb, 32), Mbp = ru(d.Ci, 64);
    const int Nbk = Bn * d.Hi * d.Wi;

    wnorm_kernel<<<d.Co, 256, 0, stream>>>(W[l], d.Ci * KK, invn);
    pack_bwd_kernel<<<(Mbp * Kbp + 255) / 256, 256, 0, stream>>>(
        W[l], invn, Apb, d.Co, d.Ci, KK, Kbp, Mbp);

    gemm(Apb, h, nullptr, P0, d.Ci, Nbk, Kbp, Kb, Mbp, 1, d.Co, d.Ho, d.Wo,
         d.Hi, d.Wi, d.K, 1.f, 0.f, 0);

    const int n  = Bn * d.Ci * d.Hi * d.Wi;
    const int HW = d.Hi * d.Wi;
    if (l > 0) {
      bn_stats_kernel<<<d.Ci, 256, 0, stream>>>(P0, Bn, d.Ci, HW, bnm, bni);
      act_kernel<<<(n + 255) / 256, 256, 0, stream>>>(P0, P1, n, d.Ci, HW,
                                                      bnm, bni, 1, 0.0f);
      float* tmp = P0; P0 = P1; P1 = tmp;
      h = P0;  // post-swap P0 is the buffer just written
    } else {
      tanh_kernel<<<(n + 255) / 256, 256, 0, stream>>>(P0, out + 8192, n);
    }
  }
}
